// LSTMClassifier_48584670052557
// MI455X (gfx1250) — matrix operand order repaired and checked by reading, not compiled, not run on hardware
//
#include <hip/hip_runtime.h>
#include <math.h>

#define B_   128
#define T_   512
#define I_   256
#define H_   512
#define G4_  2048   // 4*H
#define C_   10
#define MTOT (B_ * T_)  // 65536

typedef __bf16 bf16x16 __attribute__((ext_vector_type(16)));
typedef __bf16 bf16x8  __attribute__((ext_vector_type(8)));
typedef float  f32x8   __attribute__((ext_vector_type(8)));

union AFrag { bf16x16 v; bf16x8 h[2]; };

__device__ __forceinline__ float sigf(float x) { return 1.0f / (1.0f + __expf(-x)); }

// ---------------------------------------------------------------------------
// f32 -> bf16 conversion (grid-stride)
// ---------------------------------------------------------------------------
__global__ void cvt_f32_bf16(const float* __restrict__ in, __bf16* __restrict__ out, long n) {
  long i = (long)blockIdx.x * blockDim.x + threadIdx.x;
  long stride = (long)gridDim.x * blockDim.x;
  for (; i < n; i += stride) out[i] = (__bf16)in[i];
}

// ---------------------------------------------------------------------------
// D[orow, n] = sum_k A[m,k] * W[n,k] + bias0[n] + bias1[n]
// A: [M,K] bf16 row-major, W: [N,K] bf16 row-major, D: f32
// REMAP: input rows are indexed m = b*T + t, output rows written at t*B + b
// Block = 256 threads (8 waves); block tile 64(M) x 128(N); wave tile 16x64.
// ---------------------------------------------------------------------------
template<bool REMAP>
__global__ __launch_bounds__(256) void wmma_gemm_bias(
    const __bf16* __restrict__ A, const __bf16* __restrict__ W,
    const float* __restrict__ bias0, const float* __restrict__ bias1,
    float* __restrict__ D, int M, int N, int K) {
  const int tid  = threadIdx.x;
  const int wave = tid >> 5;
  const int lane = tid & 31;
  const int ln   = lane & 15;
  const int hi   = lane >> 4;          // 0: lanes 0-15, 1: lanes 16-31
  const int waveM = wave >> 1;         // 0..3
  const int waveN = wave & 1;          // 0..1
  const int rows = blockIdx.y * 64 + waveM * 16;
  const int cols = blockIdx.x * 128 + waveN * 64;

  f32x8 acc[4] = {};

  for (int kb = 0; kb < K; kb += 32) {
    // A fragment (16x32 bf16): lane row m=ln(+0/..), K chunks {0..15} (hi=0)
    // or {16..31} (hi=1)
    AFrag af;
    const __bf16* ap = A + (size_t)(rows + ln) * K + kb + hi * 16;
    af.h[0] = *(const bf16x8*)(ap);
    af.h[1] = *(const bf16x8*)(ap + 8);
#pragma unroll
    for (int t = 0; t < 4; ++t) {
      // B fragment (32x16 bf16): lane column n, 16 contiguous K at +16*hi
      const __bf16* wp = W + (size_t)(cols + t * 16 + ln) * K + kb + hi * 16;
      bf16x16 bf = *(const bf16x16*)wp;
      acc[t] = __builtin_amdgcn_wmma_f32_16x16x32_bf16(
          false, af.v, false, bf, (short)0, acc[t], false, false);
    }
  }

#pragma unroll
  for (int t = 0; t < 4; ++t) {
    int n = cols + t * 16 + ln;
    float bs = bias0[n] + bias1[n];
#pragma unroll
    for (int r = 0; r < 8; ++r) {
      int m = rows + r + 8 * hi;                            // C/D layout: VGPR r <-> M row
      int orow = REMAP ? ((m & (T_ - 1)) * B_ + (m >> 9))   // b*T+t -> t*B+b  (T=512)
                       : m;
      D[(size_t)orow * N + n] = acc[t][r] + bs;
    }
  }
}

// ---------------------------------------------------------------------------
// Recurrent LSTM layer. 8 blocks, one per 16-row batch tile; 1024 threads
// (32 waves). Wave w owns gate-column tiles {w, 32+w, 64+w, 96+w} = i/f/g/o
// for h columns [16w, 16w+16) -> cell update is lane-local. c state in VGPRs,
// h shared via LDS (bf16), W_hh streamed from L2 each step.
// gx: [T, B, 4H] f32.  seq_out (optional): [T, B, H] bf16.  hn (optional): [B,H] f32.
// ---------------------------------------------------------------------------
__global__ __launch_bounds__(1024) void lstm_rec(
    const float* __restrict__ gx, const __bf16* __restrict__ Whh,
    __bf16* __restrict__ seq_out, float* __restrict__ hn) {
  __shared__ __bf16 hbuf[16][H_];    // 16 KB
  const int tid  = threadIdx.x;
  const int wave = tid >> 5;         // 0..31
  const int lane = tid & 31;
  const int ln   = lane & 15;
  const int hi   = lane >> 4;
  const int rowbase = blockIdx.x * 16;
  const int colb = 16 * wave;        // h column tile base, 0..496

  for (int i = tid; i < 16 * H_; i += 1024) ((__bf16*)hbuf)[i] = (__bf16)0.0f;
  __syncthreads();

  f32x8 cst = {};                    // persistent cell state: (m=r+8*hi, j=colb+ln)

  for (int t = 0; t < T_; ++t) {
    // Initialize accumulators with the precomputed input-gate contribution
    f32x8 acc[4];
#pragma unroll
    for (int g = 0; g < 4; ++g) {
      int coln = g * H_ + colb + ln;
#pragma unroll
      for (int r = 0; r < 8; ++r) {
        int b = rowbase + r + 8 * hi;
        acc[g][r] = gx[((size_t)t * B_ + b) * G4_ + coln];
      }
    }
    // gates += h(t-1) @ W_hh^T  over K = 512
#pragma unroll 4
    for (int kb = 0; kb < H_; kb += 32) {
      AFrag af;
      const __bf16* hp = &hbuf[ln][kb + hi * 16];
      af.h[0] = *(const bf16x8*)(hp);
      af.h[1] = *(const bf16x8*)(hp + 8);
#pragma unroll
      for (int g = 0; g < 4; ++g) {
        const __bf16* wp = Whh + (size_t)(g * H_ + colb + ln) * H_ + kb + hi * 16;
        bf16x16 bf = *(const bf16x16*)wp;
        acc[g] = __builtin_amdgcn_wmma_f32_16x16x32_bf16(
            false, af.v, false, bf, (short)0, acc[g], false, false);
      }
    }
    __syncthreads();   // all waves done reading h(t-1) before overwriting
#pragma unroll
    for (int r = 0; r < 8; ++r) {
      float iv = sigf(acc[0][r]);
      float fv = sigf(acc[1][r]);
      float gv = tanhf(acc[2][r]);
      float ov = sigf(acc[3][r]);
      float cv = fv * cst[r] + iv * gv;
      cst[r] = cv;
      float hv = ov * tanhf(cv);
      int m = r + 8 * hi;
      int j = colb + ln;
      hbuf[m][j] = (__bf16)hv;
      if (seq_out) seq_out[((size_t)t * B_ + rowbase + m) * H_ + j] = (__bf16)hv;
      if (hn && t == T_ - 1) hn[(size_t)(rowbase + m) * H_ + j] = hv;
    }
    __syncthreads();   // h(t) visible before next step's fragment loads
  }
}

// ---------------------------------------------------------------------------
// logits[b,c] = hn[b,:] . fc_w[c,:] + fc_b[c]   (1.3 MFLOP - trivial)
// ---------------------------------------------------------------------------
__global__ void fc_kernel(const float* __restrict__ hn, const float* __restrict__ fcw,
                          const float* __restrict__ fcb, float* __restrict__ out) {
  int b = threadIdx.x;   // 0..127
  int c = blockIdx.x;    // 0..9
  float s = fcb[c];
  for (int k = 0; k < H_; ++k) s += hn[b * H_ + k] * fcw[c * H_ + k];
  out[b * C_ + c] = s;
}

// ---------------------------------------------------------------------------
extern "C" void kernel_launch(void* const* d_in, const int* in_sizes, int n_in,
                              void* d_out, int out_size, void* d_ws, size_t ws_size,
                              hipStream_t stream) {
  (void)in_sizes; (void)n_in; (void)out_size; (void)ws_size;
  const float* x    = (const float*)d_in[0];
  const float* Wih0 = (const float*)d_in[1];
  const float* Whh0 = (const float*)d_in[2];
  const float* bih0 = (const float*)d_in[3];
  const float* bhh0 = (const float*)d_in[4];
  const float* Wih1 = (const float*)d_in[5];
  const float* Whh1 = (const float*)d_in[6];
  const float* bih1 = (const float*)d_in[7];
  const float* bhh1 = (const float*)d_in[8];
  const float* fcw  = (const float*)d_in[9];
  const float* fcb  = (const float*)d_in[10];

  char* ws = (char*)d_ws;
  size_t off = 0;
  auto alloc = [&](size_t bytes) -> void* {
    void* p = ws + off;
    off += (bytes + 255) & ~(size_t)255;
    return p;
  };
  __bf16* xbf    = (__bf16*)alloc((size_t)MTOT * I_ * 2);   // 32 MB
  __bf16* wih0bf = (__bf16*)alloc((size_t)G4_ * I_ * 2);
  __bf16* whh0bf = (__bf16*)alloc((size_t)G4_ * H_ * 2);
  __bf16* wih1bf = (__bf16*)alloc((size_t)G4_ * H_ * 2);
  __bf16* whh1bf = (__bf16*)alloc((size_t)G4_ * H_ * 2);
  __bf16* out0bf = (__bf16*)alloc((size_t)MTOT * H_ * 2);   // 64 MB
  float*  hnbuf  = (float*)alloc((size_t)B_ * H_ * 4);
  float*  gxbuf  = (float*)alloc((size_t)MTOT * G4_ * 4);   // 512 MB (reused for both layers)

  // 1) precision conversion (one-time)
  cvt_f32_bf16<<<4096, 256, 0, stream>>>(x,    xbf,    (long)MTOT * I_);
  cvt_f32_bf16<<<2048, 256, 0, stream>>>(Wih0, wih0bf, (long)G4_ * I_);
  cvt_f32_bf16<<<2048, 256, 0, stream>>>(Whh0, whh0bf, (long)G4_ * H_);
  cvt_f32_bf16<<<2048, 256, 0, stream>>>(Wih1, wih1bf, (long)G4_ * H_);
  cvt_f32_bf16<<<2048, 256, 0, stream>>>(Whh1, whh1bf, (long)G4_ * H_);

  dim3 ggrid(G4_ / 128, MTOT / 64);  // 16 x 1024

  // 2) layer 0: hoisted input GEMM (rows b*T+t -> gx stored [T,B,4H]) + scan
  wmma_gemm_bias<true ><<<ggrid, 256, 0, stream>>>(xbf, wih0bf, bih0, bhh0, gxbuf, MTOT, G4_, I_);
  lstm_rec<<<8, 1024, 0, stream>>>(gxbuf, whh0bf, out0bf, nullptr);

  // 3) layer 1: input rows already t*B+b -> no remap; only final h needed
  wmma_gemm_bias<false><<<ggrid, 256, 0, stream>>>(out0bf, wih1bf, bih1, bhh1, gxbuf, MTOT, G4_, H_);
  lstm_rec<<<8, 1024, 0, stream>>>(gxbuf, whh1bf, nullptr, hnbuf);

  // 4) classifier head
  fc_kernel<<<C_, B_, 0, stream>>>(hnbuf, fcw, fcb, (float*)d_out);
}
